// Attention_63625645523210
// MI455X (gfx1250) — matrix operand order repaired and checked by reading, not compiled, not run on hardware
//
#include <hip/hip_runtime.h>
#include <hip/hip_bf16.h>
#include <math.h>

#define T_ 1024
#define B_ 16
#define H_ 1024
#define S_ 2048

typedef __attribute__((ext_vector_type(16))) __bf16 v16bf;
typedef __attribute__((ext_vector_type(8)))  __bf16 v8bf;
typedef __attribute__((ext_vector_type(8)))  float  v8f;

// ---------------------------------------------------------------------------
// WMMA fragment helpers (16-bit operands, contiguous k order k = half*16 + elem):
//  A 16x32: lane m=lane&15, half=lane>>4; elems 0..15 = K half*16..+15
//           (loaded as two adjacent 8-elem chunks)
//  B 32x16: lane n=lane&15, half=lane>>4; elems 0..15 = K half*16..+15
// ---------------------------------------------------------------------------
static __device__ inline v16bf frag_a_bf(const __bf16* __restrict__ p, int hf) {
  v8bf lo = *reinterpret_cast<const v8bf*>(p + hf * 16);
  v8bf hi = *reinterpret_cast<const v8bf*>(p + hf * 16 + 8);
  v16bf a;
#pragma unroll
  for (int j = 0; j < 8; ++j) { a[j] = lo[j]; a[8 + j] = hi[j]; }
  return a;
}

static __device__ inline v16bf frag_b_bf(const __bf16* __restrict__ p, int hf) {
  v8bf lo = *reinterpret_cast<const v8bf*>(p + hf * 16);
  v8bf hi = *reinterpret_cast<const v8bf*>(p + hf * 16 + 8);
  v16bf b;
#pragma unroll
  for (int j = 0; j < 8; ++j) { b[j] = lo[j]; b[8 + j] = hi[j]; }
  return b;
}

static __device__ inline v8f wmma_bf16(v16bf a, v16bf b, v8f c) {
  return __builtin_amdgcn_wmma_f32_16x16x32_bf16(false, a, false, b, (short)0, c,
                                                 false, false);
}

// LDS byte offset of a generic pointer into a __shared__ array.
static __device__ inline unsigned lds_addr_u32(const void* p) {
  return (unsigned)(unsigned long long)
      (__attribute__((address_space(3))) const char*)p;
}

// ---------------------------------------------------------------------------
// Kernel 1: fp32 -> bf16 conversion (layout preserving), 8 elems/thread
// ---------------------------------------------------------------------------
__global__ __launch_bounds__(256) void cvt_f32_bf16x8(const float* __restrict__ src,
                                                      __bf16* __restrict__ dst) {
  size_t i = ((size_t)blockIdx.x * 256 + threadIdx.x) * 8;
  v8f x = *reinterpret_cast<const v8f*>(src + i);
  v8bf y;
#pragma unroll
  for (int j = 0; j < 8; ++j) y[j] = (__bf16)x[j];
  *reinterpret_cast<v8bf*>(dst + i) = y;
}

// ---------------------------------------------------------------------------
// Kernel 2: scores[b,t,s] = sum_h out[t,b,h]*mem[s,b,h]; mask -> -inf.
// Wave computes a 16t x 64s strip (4 accumulators, shared A fragment);
// 8 waves/block -> 16t x 512s per block.
// ---------------------------------------------------------------------------
__global__ __launch_bounds__(256) void attn_scores(const __bf16* __restrict__ outbf,
                                                   const __bf16* __restrict__ membf,
                                                   const unsigned char* __restrict__ mask,
                                                   float* __restrict__ attn) {
  const int lane = threadIdx.x & 31;
  const int wave = threadIdx.x >> 5;
  const int b  = blockIdx.z;
  const int t0 = blockIdx.y * 16;
  const int sb = blockIdx.x * 512 + wave * 64;
  const int m  = lane & 15;
  const int hf = lane >> 4;
  const size_t RS = (size_t)B_ * H_;

  const __bf16* Ar = outbf + (size_t)(t0 + m) * RS + (size_t)b * H_;
  const __bf16* Bc = membf + (size_t)(sb + m) * RS + (size_t)b * H_;

  v8f acc[4] = {{}, {}, {}, {}};
#pragma unroll 2
  for (int k = 0; k < H_; k += 32) {
    __builtin_prefetch(Ar + k + 256, 0, 3);
    __builtin_prefetch(Bc + k + 256, 0, 3);
    v16bf a = frag_a_bf(Ar + k, hf);
#pragma unroll
    for (int u = 0; u < 4; ++u) {
      v16bf bb = frag_b_bf(Bc + (size_t)u * 16 * RS + k, hf);
      acc[u] = wmma_bf16(a, bb, acc[u]);
    }
  }

  // D layout: elem v -> M = v + 8*hf, N = lane&15
  float* base = attn + (size_t)b * T_ * S_;
#pragma unroll
  for (int u = 0; u < 4; ++u) {
    int s = sb + u * 16 + m;
#pragma unroll
    for (int v = 0; v < 8; ++v) {
      int t = t0 + v + 8 * hf;
      float val = acc[u][v];
      if (mask[(size_t)b * T_ * S_ + (size_t)t * S_ + s]) val = -__builtin_inff();
      base[(size_t)t * S_ + s] = val;
    }
  }
}

// ---------------------------------------------------------------------------
// Kernel 3: softmax over s (row length S=2048), in place in d_out attn region.
// ---------------------------------------------------------------------------
__global__ __launch_bounds__(256) void softmax_rows(float* __restrict__ attn) {
  __shared__ float red[256];
  const int tid = threadIdx.x;
  float* p = attn + (size_t)blockIdx.x * S_;

  float v[8];
  float mx = -__builtin_inff();
#pragma unroll
  for (int j = 0; j < 8; ++j) { v[j] = p[tid + j * 256]; mx = fmaxf(mx, v[j]); }
  red[tid] = mx; __syncthreads();
  for (int st = 128; st > 0; st >>= 1) {
    if (tid < st) red[tid] = fmaxf(red[tid], red[tid + st]);
    __syncthreads();
  }
  mx = red[0]; __syncthreads();

  float s = 0.f;
#pragma unroll
  for (int j = 0; j < 8; ++j) { v[j] = __expf(v[j] - mx); s += v[j]; }
  red[tid] = s; __syncthreads();
  for (int st = 128; st > 0; st >>= 1) {
    if (tid < st) red[tid] += red[tid + st];
    __syncthreads();
  }
  float inv = 1.0f / red[0];
#pragma unroll
  for (int j = 0; j < 8; ++j) p[tid + j * 256] = v[j] * inv;
}

// ---------------------------------------------------------------------------
// Kernel 4: context[t,b,h] = sum_s probs[b,t,s]*mem[s,b,h] (bf16 out).
// Memory columns are strided in s -> stage 32(s) x 128(h) tile in LDS via
// CDNA5 GLOBAL_LOAD_ASYNC_TO_LDS_B128 (ASYNCcnt), then read transposed.
// Block covers 32t x 128h; wave w owns h-subtile w*16 with 2 t-tiles
// sharing one B fragment (2 WMMA per staged tile per wave).
// ---------------------------------------------------------------------------
#define LDW 136  // padded LDS row stride (bf16 elems) to spread banks

__global__ __launch_bounds__(256) void attn_context(const float* __restrict__ probs,
                                                    const __bf16* __restrict__ membf,
                                                    __bf16* __restrict__ ctxbf) {
  __shared__ __bf16 lds[32 * LDW];
  const int tid  = threadIdx.x;
  const int lane = tid & 31;
  const int wave = tid >> 5;
  const int b  = blockIdx.z;
  const int t0 = blockIdx.y * 32;
  const int h0 = blockIdx.x * 128;
  const int m  = lane & 15;
  const int hf = lane >> 4;
  const size_t RS = (size_t)B_ * H_;

  const float* Prow0 = probs + ((size_t)b * T_ + (t0 + m)) * S_;
  const float* Prow1 = Prow0 + (size_t)16 * S_;
  const int sl = tid >> 3;          // 0..31 : s row of the staged tile
  const int ch = tid & 7;           // 0..7  : 16-element h chunk
  const __bf16* Msrc0 = membf + (size_t)b * H_ + h0 + ch * 16;
  const int hl = wave * 16 + m;     // h column within tile for B fragment
  const unsigned ldso = lds_addr_u32(&lds[sl * LDW + ch * 16]);

  v8f acc0 = {}, acc1 = {};
  for (int s = 0; s < S_; s += 32) {
    const __bf16* src = Msrc0 + (size_t)(s + sl) * RS;
    // 32 bytes per thread: two async b128 copies straight into LDS
    asm volatile(
        "global_load_async_to_lds_b128 %0, %1, off\n\t"
        "global_load_async_to_lds_b128 %0, %1, off offset:16"
        :
        : "v"(ldso), "v"((unsigned long long)(uintptr_t)src)
        : "memory");
    asm volatile("s_wait_asynccnt 0" ::: "memory");
    __syncthreads();

    // A fragments from fp32 probs (convert in-register)
    const float* pk0 = Prow0 + s + hf * 8;
    const float* pk1 = Prow1 + s + hf * 8;
    v8f p0lo = *reinterpret_cast<const v8f*>(pk0);
    v8f p0hi = *reinterpret_cast<const v8f*>(pk0 + 16);
    v8f p1lo = *reinterpret_cast<const v8f*>(pk1);
    v8f p1hi = *reinterpret_cast<const v8f*>(pk1 + 16);
    v16bf a0, a1;
#pragma unroll
    for (int j = 0; j < 8; ++j) {
      a0[j] = (__bf16)p0lo[j]; a0[8 + j] = (__bf16)p0hi[j];
      a1[j] = (__bf16)p1lo[j]; a1[8 + j] = (__bf16)p1hi[j];
    }

    // B fragment: transposed read from LDS (shared by both t-tiles)
    v16bf bb;
#pragma unroll
    for (int j = 0; j < 16; ++j) bb[j] = lds[(j < 8 ? hf * 8 + j : 16 + hf * 8 + (j - 8)) * LDW + hl];

    acc0 = wmma_bf16(a0, bb, acc0);
    acc1 = wmma_bf16(a1, bb, acc1);
    __syncthreads();
  }

#pragma unroll
  for (int v = 0; v < 8; ++v) {
    int t = t0 + v + 8 * hf;
    int h = h0 + wave * 16 + m;
    ctxbf[((size_t)t * B_ + b) * H_ + h] = (__bf16)acc0[v];
    ctxbf[((size_t)(t + 16) * B_ + b) * H_ + h] = (__bf16)acc1[v];
  }
}

// ---------------------------------------------------------------------------
// Kernel 5: out[r,h] = tanh( [ctx|out][r,:] . W[h,:] + bias[h] ), r = t*B+b.
// Wave computes 16r x 64h (4 accumulators, shared A fragment);
// 8 waves/block -> 16r x 512h. W columns converted fp32->bf16 in-register.
// ---------------------------------------------------------------------------
__global__ __launch_bounds__(256) void final_gemm(const __bf16* __restrict__ ctxbf,
                                                  const __bf16* __restrict__ outbf,
                                                  const float* __restrict__ W,
                                                  const float* __restrict__ bias,
                                                  float* __restrict__ out) {
  const int lane = threadIdx.x & 31;
  const int wave = threadIdx.x >> 5;
  const int r0 = blockIdx.y * 16;
  const int h0 = blockIdx.x * 512 + wave * 64;
  const int m  = lane & 15;
  const int hf = lane >> 4;

  const float* Wc = W + (size_t)(h0 + m) * (2 * H_);  // column h0+m; +u*16 rows
  const size_t WU = (size_t)16 * 2 * H_;              // 16 W-rows stride

  v8f acc[4] = {{}, {}, {}, {}};
#pragma unroll 1
  for (int phase = 0; phase < 2; ++phase) {
    const __bf16* Abase = (phase == 0 ? ctxbf : outbf) + (size_t)(r0 + m) * H_;
    const float* Wp = Wc + phase * H_;
#pragma unroll 2
    for (int k = 0; k < H_; k += 32) {
      __builtin_prefetch(Abase + k + 256, 0, 3);
      v16bf a = frag_a_bf(Abase + k, hf);
#pragma unroll
      for (int u = 0; u < 4; ++u) {
        const float* wk = Wp + u * WU + k + hf * 16;
        v8f wlo = *reinterpret_cast<const v8f*>(wk);
        v8f whi = *reinterpret_cast<const v8f*>(wk + 8);
        v16bf bb;
#pragma unroll
        for (int j = 0; j < 8; ++j) { bb[j] = (__bf16)wlo[j]; bb[8 + j] = (__bf16)whi[j]; }
        acc[u] = wmma_bf16(a, bb, acc[u]);
      }
    }
  }

#pragma unroll
  for (int u = 0; u < 4; ++u) {
    int h = h0 + u * 16 + m;
    float bv = bias[h];
#pragma unroll
    for (int v = 0; v < 8; ++v) {
      int r = r0 + v + 8 * hf;
      out[(size_t)r * H_ + h] = tanhf(acc[u][v] + bv);
    }
  }
}

// ---------------------------------------------------------------------------
// Host launcher
// ---------------------------------------------------------------------------
extern "C" void kernel_launch(void* const* d_in, const int* in_sizes, int n_in,
                              void* d_out, int out_size, void* d_ws, size_t ws_size,
                              hipStream_t stream) {
  (void)in_sizes; (void)n_in; (void)out_size; (void)ws_size;
  const float* output        = (const float*)d_in[0];   // (T,B,H) fp32
  const float* memory        = (const float*)d_in[1];   // (S,B,H) fp32
  const unsigned char* mask  = (const unsigned char*)d_in[2]; // (B,T,S) bool
  const float* W             = (const float*)d_in[3];   // (H,2H) fp32
  const float* bias          = (const float*)d_in[4];   // (H,)  fp32

  float* out  = (float*)d_out;                          // (T,B,H)
  float* attn = out + (size_t)T_ * B_ * H_;             // (B,T,S)

  __bf16* outbf = (__bf16*)d_ws;                        // T*B*H
  __bf16* membf = outbf + (size_t)T_ * B_ * H_;         // S*B*H
  __bf16* ctxbf = membf + (size_t)S_ * B_ * H_;         // T*B*H

  const size_t nOut = (size_t)T_ * B_ * H_;
  const size_t nMem = (size_t)S_ * B_ * H_;

  cvt_f32_bf16x8<<<(unsigned)(nOut / 2048), 256, 0, stream>>>(output, outbf);
  cvt_f32_bf16x8<<<(unsigned)(nMem / 2048), 256, 0, stream>>>(memory, membf);

  attn_scores<<<dim3(S_ / 512, T_ / 16, B_), 256, 0, stream>>>(outbf, membf, mask, attn);
  softmax_rows<<<B_ * T_, 256, 0, stream>>>(attn);
  attn_context<<<dim3(H_ / 128, T_ / 32, B_), 256, 0, stream>>>(attn, membf, ctxbf);
  final_gemm<<<dim3(H_ / 512, (T_ * B_) / 16), 256, 0, stream>>>(ctxbf, outbf, W, bias, out);
}
